// MoEDecoder_3908420239552
// MI455X (gfx1250) — compile-verified, matrix operand order checked by reading, not run on hardware
//
#include <hip/hip_runtime.h>
#include <hip/hip_bf16.h>

// ---------------------------------------------------------------------------
// MoE decoder (dense experts) for MI455X / gfx1250, bf16 WMMA path.
// B=8192, DIN=1024, H=2048, DOUT=1024, E=8
// ---------------------------------------------------------------------------

#define MOE_B    8192
#define MOE_DIN  1024
#define MOE_H    2048
#define MOE_DOUT 1024
#define MOE_E    8

#define BM 128
#define BN 128
#define BK 32
#define LDSK 40   // BK + 8 pad -> 80-byte row stride, keeps 16B alignment

typedef __attribute__((ext_vector_type(16))) __bf16 v16bf;
typedef __attribute__((ext_vector_type(8)))  __bf16 v8bf;
typedef __attribute__((ext_vector_type(8)))  float  v8f;

// ---------------------------------------------------------------------------
// fp32 -> bf16 elementwise convert (counts are multiples of 256)
// ---------------------------------------------------------------------------
__global__ __launch_bounds__(256) void cvt_f32_bf16(const float* __restrict__ in,
                                                    __bf16* __restrict__ out) {
  int i = blockIdx.x * 256 + threadIdx.x;
  out[i] = (__bf16)in[i];
}

// ---------------------------------------------------------------------------
// gate = softmax(x @ Wg + bg), one wave32 per row
// ---------------------------------------------------------------------------
__global__ __launch_bounds__(256) void gate_softmax(const float* __restrict__ x,
                                                    const float* __restrict__ Wg,
                                                    const float* __restrict__ bg,
                                                    float* __restrict__ gate) {
  int row  = blockIdx.x * 8 + (threadIdx.x >> 5);
  int lane = threadIdx.x & 31;
  const float* xr = x + (size_t)row * MOE_DIN;
  float acc[MOE_E];
#pragma unroll
  for (int e = 0; e < MOE_E; ++e) acc[e] = 0.f;
  for (int k = lane; k < MOE_DIN; k += 32) {
    float xv = xr[k];
    const float* w = Wg + (size_t)k * MOE_E;
#pragma unroll
    for (int e = 0; e < MOE_E; ++e) acc[e] += xv * w[e];
  }
#pragma unroll
  for (int e = 0; e < MOE_E; ++e)
#pragma unroll
    for (int off = 16; off > 0; off >>= 1)
      acc[e] += __shfl_xor(acc[e], off, 32);
  if (lane == 0) {
    float mx = -1e30f;
#pragma unroll
    for (int e = 0; e < MOE_E; ++e) { acc[e] += bg[e]; mx = fmaxf(mx, acc[e]); }
    float s = 0.f;
#pragma unroll
    for (int e = 0; e < MOE_E; ++e) { acc[e] = __expf(acc[e] - mx); s += acc[e]; }
    float inv = 1.f / s;
#pragma unroll
    for (int e = 0; e < MOE_E; ++e) gate[(size_t)row * MOE_E + e] = acc[e] * inv;
  }
}

// ---------------------------------------------------------------------------
// mish + LayerNorm over last dim (H=2048), one 256-thread block per row.
// reads fp32 pre-activation, writes bf16 activation.
// ---------------------------------------------------------------------------
__global__ __launch_bounds__(256) void mish_ln(const float* __restrict__ h,
                                               __bf16* __restrict__ act) {
  int row = blockIdx.x;
  const float* hr = h + (size_t)row * MOE_H;
  __bf16* ar = act + (size_t)row * MOE_H;

  float vals[MOE_H / 256];
  float s = 0.f, ss = 0.f;
#pragma unroll
  for (int i = 0; i < MOE_H / 256; ++i) {
    float v  = hr[threadIdx.x + i * 256];
    float sp = (v > 20.f) ? v : log1pf(__expf(v));   // softplus, overflow-safe
    float m  = v * tanhf(sp);                        // mish
    vals[i] = m;  s += m;  ss += m * m;
  }
  __shared__ float red[2][8];
#pragma unroll
  for (int off = 16; off > 0; off >>= 1) {
    s  += __shfl_down(s,  off, 32);
    ss += __shfl_down(ss, off, 32);
  }
  int wid = threadIdx.x >> 5, lane = threadIdx.x & 31;
  if (lane == 0) { red[0][wid] = s; red[1][wid] = ss; }
  __syncthreads();
  if (threadIdx.x == 0) {
    float a = 0.f, b = 0.f;
#pragma unroll
    for (int i = 0; i < 8; ++i) { a += red[0][i]; b += red[1][i]; }
    red[0][0] = a * (1.f / MOE_H);
    red[1][0] = b * (1.f / MOE_H);
  }
  __syncthreads();
  float mu  = red[0][0];
  float var = red[1][0] - mu * mu;
  float rstd = rsqrtf(var + 1e-5f);
#pragma unroll
  for (int i = 0; i < MOE_H / 256; ++i)
    ar[threadIdx.x + i * 256] = (__bf16)((vals[i] - mu) * rstd);
}

// ---------------------------------------------------------------------------
// bf16 WMMA GEMM: C[M,N] = A[M,K] @ W[K,N] + bias   (fp32 out)
// 256 threads = 8 waves; BM=128 BN=128 BK=32; per wave 32x64 -> 2x4 WMMA tiles.
// FINAL: out[b,n] (+)= gate[b,e] * (acc + bias[n])
// ---------------------------------------------------------------------------
template<bool FINAL>
__global__ __launch_bounds__(256)
void gemm_bf16_wmma(const __bf16* __restrict__ A, const __bf16* __restrict__ W,
                    const float* __restrict__ bias, float* __restrict__ C,
                    int M, int N, int K,
                    const float* __restrict__ gate, int expert, int first) {
  __shared__ __attribute__((aligned(16))) __bf16 sA[2][BM][LDSK];
  __shared__ __attribute__((aligned(16))) __bf16 sB[2][BN][LDSK];  // [n][k] (transposed)

  const int tid  = threadIdx.x;
  const int lane = tid & 31;
  const int wid  = tid >> 5;
  const int wm   = wid & 3;   // wave M offset: wm*32
  const int wn   = wid >> 2;  // wave N offset: wn*64
  const int rowBase = blockIdx.y * BM;
  const int colBase = blockIdx.x * BN;

  v8f acc[2][4] = {};

  // A tile: 128x32, 512 chunks of 8 along K.   chunk c: row=c>>2, kcol=(c&3)*8
  // W tile: 32x128, 512 chunks of 8 along N.   chunk c: k=c>>4,   ncol=(c&15)*8
  auto gA = [&](int kt, int c) -> v8bf {
    int r = c >> 2, kc = (c & 3) * 8;
    return *(const v8bf*)(A + (size_t)(rowBase + r) * K + kt * BK + kc);
  };
  auto gB = [&](int kt, int c) -> v8bf {
    int k = c >> 4, nc = (c & 15) * 8;
    return *(const v8bf*)(W + (size_t)(kt * BK + k) * N + colBase + nc);
  };
  auto stage = [&](int buf, v8bf a0, v8bf a1, v8bf b0, v8bf b1) {
    int c = tid;
    *(v8bf*)&sA[buf][c >> 2][(c & 3) * 8] = a0;
    c = tid + 256;
    *(v8bf*)&sA[buf][c >> 2][(c & 3) * 8] = a1;
    c = tid;
    {
      int k = c >> 4, nc = (c & 15) * 8;
#pragma unroll
      for (int j = 0; j < 8; ++j) sB[buf][nc + j][k] = b0[j];
    }
    c = tid + 256;
    {
      int k = c >> 4, nc = (c & 15) * 8;
#pragma unroll
      for (int j = 0; j < 8; ++j) sB[buf][nc + j][k] = b1[j];
    }
  };

  const int KT = K / BK;

  v8bf a0 = gA(0, tid), a1 = gA(0, tid + 256);
  v8bf b0 = gB(0, tid), b1 = gB(0, tid + 256);
  stage(0, a0, a1, b0, b1);
  __syncthreads();

  const int rsel = (lane >> 4) * 8;  // K half-select per documented 16-bit A layout
  const int rrow = lane & 15;

  for (int kt = 0; kt < KT; ++kt) {
    const int cur = kt & 1, nxt = cur ^ 1;
    if (kt + 1 < KT) {
      a0 = gA(kt + 1, tid); a1 = gA(kt + 1, tid + 256);
      b0 = gB(kt + 1, tid); b1 = gB(kt + 1, tid + 256);
    }
    if (kt + 2 < KT) {  // global_prefetch_b8 for the tile after next
      __builtin_prefetch(A + (size_t)(rowBase + (tid >> 1)) * K + (kt + 2) * BK, 0, 1);
      __builtin_prefetch(W + (size_t)((kt + 2) * BK + (tid >> 4)) * N + colBase, 0, 1);
    }

    v16bf afrag[2], bfrag[4];
#pragma unroll
    for (int im = 0; im < 2; ++im) {
      const __bf16* p = &sA[cur][wm * 32 + im * 16 + rrow][0];
      v8bf lo = *(const v8bf*)(p + rsel);
      v8bf hi = *(const v8bf*)(p + 16 + rsel);
      afrag[im] = __builtin_shufflevector(lo, hi, 0, 1, 2, 3, 4, 5, 6, 7,
                                          8, 9, 10, 11, 12, 13, 14, 15);
    }
#pragma unroll
    for (int in = 0; in < 4; ++in) {
      const __bf16* p = &sB[cur][wn * 64 + in * 16 + rrow][0];
      v8bf lo = *(const v8bf*)(p + rsel);
      v8bf hi = *(const v8bf*)(p + 16 + rsel);
      bfrag[in] = __builtin_shufflevector(lo, hi, 0, 1, 2, 3, 4, 5, 6, 7,
                                          8, 9, 10, 11, 12, 13, 14, 15);
    }
#pragma unroll
    for (int im = 0; im < 2; ++im)
#pragma unroll
      for (int in = 0; in < 4; ++in)
        acc[im][in] = __builtin_amdgcn_wmma_f32_16x16x32_bf16(
            false, afrag[im], false, bfrag[in], (short)0, acc[im][in],
            false, false);

    if (kt + 1 < KT) stage(nxt, a0, a1, b0, b1);
    __syncthreads();
  }

  // Epilogue: C/D layout — VGPR r: lanes 0-15 -> M=r, lanes 16-31 -> M=8+r; N=lane&15
  const int mh = (lane >> 4) * 8;
  const int nn = lane & 15;
#pragma unroll
  for (int im = 0; im < 2; ++im) {
#pragma unroll
    for (int in = 0; in < 4; ++in) {
      int col = colBase + wn * 64 + in * 16 + nn;
      float bv = bias[col];
#pragma unroll
      for (int r = 0; r < 8; ++r) {
        int row = rowBase + wm * 32 + im * 16 + mh + r;
        float v = acc[im][in][r] + bv;
        size_t idx = (size_t)row * N + col;
        if (FINAL) {
          float g = gate[(size_t)row * MOE_E + expert];
          float prev = first ? 0.f : C[idx];
          C[idx] = prev + g * v;
        } else {
          C[idx] = v;
        }
      }
    }
  }
}

// ---------------------------------------------------------------------------
// host-side orchestration
// ---------------------------------------------------------------------------
static inline size_t align256(size_t x) { return (x + 255) & ~(size_t)255; }

extern "C" void kernel_launch(void* const* d_in, const int* in_sizes, int n_in,
                              void* d_out, int out_size, void* d_ws, size_t ws_size,
                              hipStream_t stream) {
  const float* x  = (const float*)d_in[0];
  const float* Wg = (const float*)d_in[1];
  const float* bg = (const float*)d_in[2];
  const float* W1 = (const float*)d_in[3];
  const float* b1 = (const float*)d_in[4];
  const float* W2 = (const float*)d_in[5];
  const float* b2 = (const float*)d_in[6];
  const float* W3 = (const float*)d_in[7];
  const float* b3 = (const float*)d_in[8];
  const float* W4 = (const float*)d_in[9];
  const float* b4 = (const float*)d_in[10];
  const float* W5 = (const float*)d_in[11];
  const float* b5 = (const float*)d_in[12];
  float* out = (float*)d_out;

  // workspace layout
  char* ws = (char*)d_ws;
  size_t off = 0;
  float*  gate = (float*)(ws + off);  off = align256(off + (size_t)MOE_B * MOE_E * 4);
  __bf16* xbf  = (__bf16*)(ws + off); off = align256(off + (size_t)MOE_B * MOE_DIN * 2);
  __bf16* wbf  = (__bf16*)(ws + off); off = align256(off + (size_t)MOE_H * MOE_H * 2);
  float*  hbuf = (float*)(ws + off);  off = align256(off + (size_t)MOE_B * MOE_H * 4);
  __bf16* act  = (__bf16*)(ws + off); off = align256(off + (size_t)MOE_B * MOE_H * 2);
  (void)ws_size; (void)n_in; (void)in_sizes; (void)out_size;

  // 1) gate
  gate_softmax<<<MOE_B / 8, 256, 0, stream>>>(x, Wg, bg, gate);
  // 2) x -> bf16
  cvt_f32_bf16<<<(MOE_B * MOE_DIN) / 256, 256, 0, stream>>>(x, xbf);

  const dim3 blk(256);
  const dim3 gridH(MOE_H / BN, MOE_B / BM);    // N=2048 layers
  const dim3 gridD(MOE_DOUT / BN, MOE_B / BM); // N=1024 final layer

  for (int e = 0; e < MOE_E; ++e) {
    // ---- layer 1: [B,DIN] @ [DIN,H] ----
    cvt_f32_bf16<<<(MOE_DIN * MOE_H) / 256, blk, 0, stream>>>(
        W1 + (size_t)e * MOE_DIN * MOE_H, wbf);
    gemm_bf16_wmma<false><<<gridH, blk, 0, stream>>>(
        xbf, wbf, b1 + (size_t)e * MOE_H, hbuf,
        MOE_B, MOE_H, MOE_DIN, nullptr, 0, 0);
    mish_ln<<<MOE_B, blk, 0, stream>>>(hbuf, act);

    // ---- layers 2..4: [B,H] @ [H,H] ----
    const float* Wm[3] = {W2, W3, W4};
    const float* bm[3] = {b2, b3, b4};
    for (int l = 0; l < 3; ++l) {
      cvt_f32_bf16<<<(MOE_H * MOE_H) / 256, blk, 0, stream>>>(
          Wm[l] + (size_t)e * MOE_H * MOE_H, wbf);
      gemm_bf16_wmma<false><<<gridH, blk, 0, stream>>>(
          act, wbf, bm[l] + (size_t)e * MOE_H, hbuf,
          MOE_B, MOE_H, MOE_H, nullptr, 0, 0);
      mish_ln<<<MOE_B, blk, 0, stream>>>(hbuf, act);
    }

    // ---- layer 5 + gated accumulate into d_out ----
    cvt_f32_bf16<<<(MOE_H * MOE_DOUT) / 256, blk, 0, stream>>>(
        W5 + (size_t)e * MOE_H * MOE_DOUT, wbf);
    gemm_bf16_wmma<true><<<gridD, blk, 0, stream>>>(
        act, wbf, b5 + (size_t)e * MOE_DOUT, out,
        MOE_B, MOE_DOUT, MOE_H, gate, e, e == 0 ? 1 : 0);
  }
}
